// MPNNFP_54494545052140
// MI455X (gfx1250) — matrix operand order repaired and checked by reading, not compiled, not run on hardware
//
#include <hip/hip_runtime.h>
#include <hip/hip_bf16.h>
#include <math.h>

// ---------------- problem constants ----------------
#define DIMD     256
#define NNODES   2000
#define NEDGES   4000
#define NGRAPHS  64
#define MOLIN    15
#define EDGEIN   4
#define FPDIM    1024
#define FPLIN    256
#define KHID     128               // edge MLP hidden
#define KTOT     (KHID * DIMD)     // 32768: virtual K of fused outer-product GEMM
#define BN_EPS   1e-5f

typedef __attribute__((ext_vector_type(16))) __bf16 v16bf;
typedef __attribute__((ext_vector_type(8)))  float  v8f;

static __device__ __forceinline__ __bf16 f2bf(float f) { return (__bf16)f; }
static __device__ __forceinline__ float sigmoidf(float x) { return 1.f / (1.f + __expf(-x)); }

// async copy 16B global -> LDS, tracked by ASYNCcnt (CDNA5 ISA)
static __device__ __forceinline__ void async_b128(unsigned lds_off, const void* gaddr) {
    asm volatile("global_load_async_to_lds_b128 %0, %1, off"
                 :: "v"(lds_off), "v"((unsigned long long)(uintptr_t)gaddr) : "memory");
}
static __device__ __forceinline__ void wait_async0() {
    asm volatile("s_wait_asynccnt 0x0" ::: "memory");
}

// =====================================================================
// Generic bf16 WMMA GEMM:  C = act( A(f32)[MxK] * B(f32) (+bias) (+beta*C) )
// Block: 128 threads = 4 waves, each wave owns one 16x16 N-subtile.
// grid = (M/16, N/64).  REQUIRES: M%16==0, K%32==0, grid tiles N exactly.
// TRANSB: B stored [N,K] row-major (torch nn.Linear weight layout).
// =====================================================================
template <int TRANSB>
__global__ void __launch_bounds__(128)
wmma_gemm_bf16(const float* __restrict__ A, const float* __restrict__ B,
               const float* __restrict__ bias, float* __restrict__ C,
               int K, int lda, int ldb, int ldc, int beta, int relu)
{
    const int lane = threadIdx.x & 31;
    const int wave = threadIdx.x >> 5;
    const int g    = lane >> 4;
    const int l16  = lane & 15;
    const int mb   = blockIdx.x * 16;
    const int nb   = (blockIdx.y * 4 + wave) * 16;
    const int bcol = nb + l16;

    const float* ap = A + (size_t)(mb + l16) * lda;     // lane's A row
    v8f acc = {0.f, 0.f, 0.f, 0.f, 0.f, 0.f, 0.f, 0.f};

    for (int kb = 0; kb < K; kb += 32) {
        v16bf a, b;
        // A fragment: rows = lane&15; K = e + 16g (16 contiguous K)
        const float* ar = ap + kb + 16 * g;
        float4 a0 = *(const float4*)(ar);
        float4 a1 = *(const float4*)(ar + 4);
        float4 a2 = *(const float4*)(ar + 8);
        float4 a3 = *(const float4*)(ar + 12);
        a[0]  = f2bf(a0.x); a[1]  = f2bf(a0.y); a[2]  = f2bf(a0.z); a[3]  = f2bf(a0.w);
        a[4]  = f2bf(a1.x); a[5]  = f2bf(a1.y); a[6]  = f2bf(a1.z); a[7]  = f2bf(a1.w);
        a[8]  = f2bf(a2.x); a[9]  = f2bf(a2.y); a[10] = f2bf(a2.z); a[11] = f2bf(a2.w);
        a[12] = f2bf(a3.x); a[13] = f2bf(a3.y); a[14] = f2bf(a3.z); a[15] = f2bf(a3.w);
        // B fragment: cols = lane&15; K = e + 16g
        if (TRANSB) {
            const float* br = B + (size_t)bcol * ldb + kb + 16 * g;   // 16 contiguous K
            float4 b0 = *(const float4*)(br);
            float4 b1 = *(const float4*)(br + 4);
            float4 b2 = *(const float4*)(br + 8);
            float4 b3 = *(const float4*)(br + 12);
            b[0]  = f2bf(b0.x); b[1]  = f2bf(b0.y); b[2]  = f2bf(b0.z); b[3]  = f2bf(b0.w);
            b[4]  = f2bf(b1.x); b[5]  = f2bf(b1.y); b[6]  = f2bf(b1.z); b[7]  = f2bf(b1.w);
            b[8]  = f2bf(b2.x); b[9]  = f2bf(b2.y); b[10] = f2bf(b2.z); b[11] = f2bf(b2.w);
            b[12] = f2bf(b3.x); b[13] = f2bf(b3.y); b[14] = f2bf(b3.z); b[15] = f2bf(b3.w);
        } else {
#pragma unroll
            for (int e = 0; e < 16; ++e)
                b[e] = f2bf(B[(size_t)(kb + e + 16 * g) * ldb + bcol]);
        }
        acc = __builtin_amdgcn_wmma_f32_16x16x32_bf16(false, a, false, b,
                                                      (short)0, acc, false, false);
    }
#pragma unroll
    for (int r = 0; r < 8; ++r) {
        float v = acc[r];
        if (bias) v += bias[bcol];
        size_t idx = (size_t)(mb + r + 8 * g) * ldc + bcol;
        if (beta) v += C[idx];
        if (relu && v < 0.f) v = 0.f;
        C[idx] = v;
    }
}

// =====================================================================
// Fused message GEMM:
//   msg[e,o] = sum_{k,i} he[e,k]*xsrc[e,i]*W2[k,i*256+o] + sum_i xsrc[e,i]*b2[i*256+o]
// A[E x (32768+256)] generated on the fly in LDS; B = w2|b2 viewed [.,256],
// staged per 32-row chunk via global_load_async_to_lds_b128 (ASYNCcnt).
// Block: 256 thr = 8 waves; 64 edges/block; wave = 2 N-tiles x 4 M-tiles.
// =====================================================================
__global__ void __launch_bounds__(256)
msg_outer_gemm(const float* __restrict__ xsrc,   // [E,256]
               const float* __restrict__ he,     // [E,128]
               const float* __restrict__ w2,     // [128,65536] == [32768,256]
               const float* __restrict__ b2,     // [65536]      == [256,256]
               float* __restrict__ msg)          // [E,256] (pure store)
{
    __shared__ float xs[64 * 256];      // 64 KB
    __shared__ float hs[64 * 128];      // 32 KB
    __shared__ float bshf[32 * 256];    // 32 KB  (128 KB total)

    const int t  = threadIdx.x;
    const int eb = blockIdx.x * 64;

    // stage xsrc tile (4096 float4)
#pragma unroll
    for (int v = 0; v < 16; ++v) {
        int i4 = v * 256 + t;
        int e = i4 >> 6, c4 = i4 & 63;
        float4 val = make_float4(0.f, 0.f, 0.f, 0.f);
        if (eb + e < NEDGES) val = ((const float4*)xsrc)[(size_t)(eb + e) * 64 + c4];
        ((float4*)xs)[i4] = val;
    }
    // stage he tile (2048 float4)
#pragma unroll
    for (int v = 0; v < 8; ++v) {
        int i4 = v * 256 + t;
        int e = i4 >> 5, c4 = i4 & 31;
        float4 val = make_float4(0.f, 0.f, 0.f, 0.f);
        if (eb + e < NEDGES) val = ((const float4*)he)[(size_t)(eb + e) * 32 + c4];
        ((float4*)hs)[i4] = val;
    }

    const int lane = t & 31;
    const int wave = t >> 5;
    const int g    = lane >> 4;
    const int l16  = lane & 15;
    const unsigned lds_b = (unsigned)(uintptr_t)&bshf[0];   // low 32 bits = LDS offset

    v8f acc[2][4];
#pragma unroll
    for (int s = 0; s < 2; ++s)
#pragma unroll
        for (int mt = 0; mt < 4; ++mt)
            acc[s][mt] = (v8f){0.f, 0.f, 0.f, 0.f, 0.f, 0.f, 0.f, 0.f};

    // main K loop (32768 outer-product K) + 8 folded bias chunks (K=256)
    for (int kb = 0; kb < KTOT + DIMD; kb += 32) {
        const int  isbias = (kb >= KTOT);
        const float* bsrc = isbias ? (b2 + (size_t)(kb - KTOT) * 256)
                                   : (w2 + (size_t)kb * 256);
        __syncthreads();                  // previous chunk fully consumed
        // async-stage B chunk [32x256] f32 (8 x b128 per thread, coalesced rows)
#pragma unroll
        for (int v = 0; v < 8; ++v) {
            int i4 = v * 256 + t;
            async_b128(lds_b + (unsigned)i4 * 16u, bsrc + (size_t)i4 * 4);
        }
        wait_async0();                    // this wave's LDS writes landed
        __syncthreads();                  // all waves' writes landed

        const int   xbase = isbias ? (kb - KTOT) : (kb & 255);
        const int   kq    = kb >> 8;      // he column (main pass only)

        // 4 A fragments (one per 16-edge M-subtile), reused by both N-tiles
        v16bf a[4];
#pragma unroll
        for (int mt = 0; mt < 4; ++mt) {
            int mrow = mt * 16 + l16;
            float hv = isbias ? 1.f : hs[mrow * 128 + kq];
            const float* xr = &xs[mrow * 256 + xbase + 16 * g];
            float4 x0 = *(const float4*)(xr);
            float4 x1 = *(const float4*)(xr + 4);
            float4 x2 = *(const float4*)(xr + 8);
            float4 x3 = *(const float4*)(xr + 12);
            a[mt][0]  = f2bf(hv * x0.x); a[mt][1]  = f2bf(hv * x0.y);
            a[mt][2]  = f2bf(hv * x0.z); a[mt][3]  = f2bf(hv * x0.w);
            a[mt][4]  = f2bf(hv * x1.x); a[mt][5]  = f2bf(hv * x1.y);
            a[mt][6]  = f2bf(hv * x1.z); a[mt][7]  = f2bf(hv * x1.w);
            a[mt][8]  = f2bf(hv * x2.x); a[mt][9]  = f2bf(hv * x2.y);
            a[mt][10] = f2bf(hv * x2.z); a[mt][11] = f2bf(hv * x2.w);
            a[mt][12] = f2bf(hv * x3.x); a[mt][13] = f2bf(hv * x3.y);
            a[mt][14] = f2bf(hv * x3.z); a[mt][15] = f2bf(hv * x3.w);
        }
#pragma unroll
        for (int s = 0; s < 2; ++s) {
            int n = wave * 32 + s * 16 + l16;
            v16bf b;
#pragma unroll
            for (int e = 0; e < 16; ++e) b[e] = f2bf(bshf[(e + 16 * g) * 256 + n]);
#pragma unroll
            for (int mt = 0; mt < 4; ++mt)
                acc[s][mt] = __builtin_amdgcn_wmma_f32_16x16x32_bf16(
                    false, a[mt], false, b, (short)0, acc[s][mt], false, false);
        }
    }

    // pure stores; block-uniform fast path for full 64-edge tiles
    if (eb + 64 <= NEDGES) {
#pragma unroll
        for (int s = 0; s < 2; ++s) {
            int n = wave * 32 + s * 16 + l16;
#pragma unroll
            for (int mt = 0; mt < 4; ++mt)
#pragma unroll
                for (int r = 0; r < 8; ++r)
                    msg[(size_t)(eb + mt * 16 + r + 8 * g) * DIMD + n] = acc[s][mt][r];
        }
    } else {
#pragma unroll
        for (int s = 0; s < 2; ++s) {
            int n = wave * 32 + s * 16 + l16;
#pragma unroll
            for (int mt = 0; mt < 4; ++mt)
#pragma unroll
                for (int r = 0; r < 8; ++r) {
                    int e = eb + mt * 16 + r + 8 * g;
                    if (e < NEDGES) msg[(size_t)e * DIMD + n] = acc[s][mt][r];
                }
        }
    }
}

// ---------------- small VALU kernels ----------------
__global__ void k_zero(float* p, int n) {
    int i = blockIdx.x * blockDim.x + threadIdx.x;
    if (i < n) p[i] = 0.f;
}

__global__ void k_lin0(const float* __restrict__ x, const float* __restrict__ w,
                       const float* __restrict__ b, float* __restrict__ out) {
    int i = blockIdx.x * blockDim.x + threadIdx.x;        // 2000*256
    if (i >= NNODES * DIMD) return;
    int n = i >> 8, c = i & 255;
    float s = b[c];
#pragma unroll
    for (int k = 0; k < MOLIN; ++k) s += x[n * MOLIN + k] * w[k * DIMD + c];
    out[i] = fmaxf(s, 0.f);
}

__global__ void k_edge1(const float* __restrict__ ea, const float* __restrict__ w,
                        const float* __restrict__ b, float* __restrict__ he) {
    int i = blockIdx.x * blockDim.x + threadIdx.x;        // 4000*128
    if (i >= NEDGES * KHID) return;
    int e = i >> 7, k = i & 127;
    float s = b[k];
#pragma unroll
    for (int j = 0; j < EDGEIN; ++j) s += ea[e * EDGEIN + j] * w[j * KHID + k];
    he[i] = fmaxf(s, 0.f);
}

__global__ void k_bnelu(const float* __restrict__ h, const float* __restrict__ gmm,
                        const float* __restrict__ bta, float* __restrict__ out) {
    int i = blockIdx.x * blockDim.x + threadIdx.x;        // 64*256
    if (i >= NGRAPHS * FPLIN) return;
    int c = i & 255;
    float v = h[i] * (gmm[c] * rsqrtf(1.f + BN_EPS)) + bta[c];
    out[i] = v > 0.f ? v : (__expf(v) - 1.f);
}

__global__ void k_deg(const int* __restrict__ dst, float* __restrict__ deg) {
    int e = blockIdx.x * blockDim.x + threadIdx.x;
    if (e < NEDGES) atomicAdd(&deg[dst[e]], 1.f);
}

__global__ void k_recip(float* d) {
    int n = blockIdx.x * blockDim.x + threadIdx.x;
    if (n < NNODES) d[n] = 1.f / fmaxf(d[n], 1.f);
}

__global__ void k_gather(const float* __restrict__ h, const int* __restrict__ src,
                         float* __restrict__ xsrc) {
    int i = blockIdx.x * blockDim.x + threadIdx.x;        // 4000*64 float4
    if (i >= NEDGES * 64) return;
    int e = i >> 6, c4 = i & 63;
    ((float4*)xsrc)[i] = ((const float4*)h)[(size_t)src[e] * 64 + c4];
}

__global__ void k_scatter(const float* __restrict__ msg, const int* __restrict__ dst,
                          float* __restrict__ agg) {
    int i = blockIdx.x * blockDim.x + threadIdx.x;        // 4000*256
    if (i >= NEDGES * DIMD) return;
    int e = i >> 8, c = i & 255;
    atomicAdd(&agg[(size_t)dst[e] * DIMD + c], msg[i]);
}

__global__ void k_combine(const float* __restrict__ agg, const float* __restrict__ rdeg,
                          const float* __restrict__ bias, float* __restrict__ m) {
    int i = blockIdx.x * blockDim.x + threadIdx.x;        // 2000*256
    if (i >= NNODES * DIMD) return;
    int n = i >> 8, c = i & 255;
    float v = agg[i] * rdeg[n] + m[i] + bias[c];          // m holds out@conv_root
    m[i] = fmaxf(v, 0.f);
}

__global__ void k_gru(const float* __restrict__ gi, const float* __restrict__ gh,
                      float* __restrict__ h) {
    int i = blockIdx.x * blockDim.x + threadIdx.x;        // 2000*256
    if (i >= NNODES * DIMD) return;
    int n = i >> 8, c = i & 255;
    size_t b = (size_t)n * 768;
    float r  = sigmoidf(gi[b + c] + gh[b + c]);
    float z  = sigmoidf(gi[b + 256 + c] + gh[b + 256 + c]);
    float nn = tanhf(gi[b + 512 + c] + r * gh[b + 512 + c]);
    h[i] = (1.f - z) * nn + z * h[i];
}

__global__ void k_lstm(const float* __restrict__ gates, float* __restrict__ lh,
                       float* __restrict__ lc, float* __restrict__ qstar) {
    int i = blockIdx.x * blockDim.x + threadIdx.x;        // 64*256
    if (i >= NGRAPHS * DIMD) return;
    int gph = i >> 8, c = i & 255;
    size_t b = (size_t)gph * 1024;
    float ig = sigmoidf(gates[b + c]);
    float fg = sigmoidf(gates[b + 256 + c]);
    float gg = tanhf(gates[b + 512 + c]);
    float og = sigmoidf(gates[b + 768 + c]);
    float cc = fg * lc[i] + ig * gg;
    lc[i] = cc;
    float hh = og * tanhf(cc);
    lh[i] = hh;
    qstar[(size_t)gph * 512 + c] = hh;                    // q part of q_star
}

__global__ void k_attn_dot(const float* __restrict__ h, const float* __restrict__ q,
                           const int* __restrict__ batch, float* __restrict__ e) {
    int n = blockIdx.x * blockDim.x + threadIdx.x;
    if (n >= NNODES) return;
    const float* hr = h + (size_t)n * DIMD;
    const float* qr = q + (size_t)batch[n] * DIMD;
    float s = 0.f;
    for (int c = 0; c < DIMD; ++c) s += hr[c] * qr[c];
    e[n] = s;
}

__global__ void k_attn_norm(const float* __restrict__ e, const int* __restrict__ batch,
                            float* __restrict__ a) {
    __shared__ float red[256];
    int gph = blockIdx.x, tid = threadIdx.x;
    float mx = -3.402823e38f;
    for (int n = tid; n < NNODES; n += 256)
        if (batch[n] == gph) mx = fmaxf(mx, e[n]);
    red[tid] = mx; __syncthreads();
    for (int s = 128; s > 0; s >>= 1) {
        if (tid < s) red[tid] = fmaxf(red[tid], red[tid + s]);
        __syncthreads();
    }
    float emax = red[0]; __syncthreads();
    float sm = 0.f;
    for (int n = tid; n < NNODES; n += 256)
        if (batch[n] == gph) sm += __expf(e[n] - emax);
    red[tid] = sm; __syncthreads();
    for (int s = 128; s > 0; s >>= 1) {
        if (tid < s) red[tid] += red[tid + s];
        __syncthreads();
    }
    float esum = fmaxf(red[0], 1e-30f);
    for (int n = tid; n < NNODES; n += 256)
        if (batch[n] == gph) a[n] = __expf(e[n] - emax) / esum;
}

__global__ void k_attn_r(const float* __restrict__ a, const float* __restrict__ h,
                         const int* __restrict__ batch, float* __restrict__ qstar) {
    int gph = blockIdx.x, c = threadIdx.x;
    float r = 0.f;
    for (int n = 0; n < NNODES; ++n)
        if (batch[n] == gph) r += a[n] * h[(size_t)n * DIMD + c];
    qstar[(size_t)gph * 512 + 256 + c] = r;               // r part of q_star
}

__global__ void k_head(const float* __restrict__ o, const float* __restrict__ ofp,
                       const float* __restrict__ w2, const float* __restrict__ b2,
                       float* __restrict__ out) {
    int i = threadIdx.x;                                  // 128 = 64*2
    if (i >= NGRAPHS * 2) return;
    int gph = i >> 1, oc = i & 1;
    float s = b2[oc];
    for (int c = 0; c < 256; ++c) s += o[(size_t)gph * 256 + c]   * w2[c * 2 + oc];
    for (int c = 0; c < 256; ++c) s += ofp[(size_t)gph * 256 + c] * w2[(256 + c) * 2 + oc];
    out[i] = s;
}

// ---------------- workspace carving ----------------
static inline float* carve(void* ws, size_t& off, size_t nfloats) {
    float* p = (float*)((char*)ws + off);
    off += (nfloats * sizeof(float) + 255) & ~(size_t)255;
    return p;
}

extern "C" void kernel_launch(void* const* d_in, const int* in_sizes, int n_in,
                              void* d_out, int out_size, void* d_ws, size_t ws_size,
                              hipStream_t stream) {
    const float* x         = (const float*)d_in[0];
    const int*   eindex    = (const int*)  d_in[1];
    const float* eattr     = (const float*)d_in[2];
    const int*   batch     = (const int*)  d_in[3];
    const float* fp        = (const float*)d_in[4];
    const float* fc1_w     = (const float*)d_in[5];
    const float* fc1_b     = (const float*)d_in[6];
    const float* bn1_g     = (const float*)d_in[7];
    const float* bn1_b     = (const float*)d_in[8];
    const float* lin0_w    = (const float*)d_in[9];
    const float* lin0_b    = (const float*)d_in[10];
    const float* emlp_w1   = (const float*)d_in[11];
    const float* emlp_b1   = (const float*)d_in[12];
    const float* emlp_w2   = (const float*)d_in[13];
    const float* emlp_b2   = (const float*)d_in[14];
    const float* conv_root = (const float*)d_in[15];
    const float* conv_bias = (const float*)d_in[16];
    const float* gru_w_ih  = (const float*)d_in[17];
    const float* gru_w_hh  = (const float*)d_in[18];
    const float* gru_b_ih  = (const float*)d_in[19];
    const float* gru_b_hh  = (const float*)d_in[20];
    const float* lstm_w_ih = (const float*)d_in[21];
    const float* lstm_w_hh = (const float*)d_in[22];
    const float* lstm_b_ih = (const float*)d_in[23];
    const float* lstm_b_hh = (const float*)d_in[24];
    const float* lin1_w    = (const float*)d_in[25];
    const float* lin1_b    = (const float*)d_in[26];
    const float* lin2_w    = (const float*)d_in[27];
    const float* lin2_b    = (const float*)d_in[28];
    const int* src = eindex;
    const int* dst = eindex + NEDGES;

    size_t off = 0;
    float* hbuf  = carve(d_ws, off, (size_t)NNODES * DIMD);   // out == h
    float* outfp = carve(d_ws, off, (size_t)NGRAPHS * FPLIN);
    float* fptmp = carve(d_ws, off, (size_t)NGRAPHS * FPLIN);
    float* he    = carve(d_ws, off, (size_t)NEDGES * KHID);
    float* xsrc  = carve(d_ws, off, (size_t)NEDGES * DIMD);
    float* msg   = carve(d_ws, off, (size_t)NEDGES * DIMD);
    float* agg   = carve(d_ws, off, (size_t)NNODES * DIMD);
    float* mbuf  = carve(d_ws, off, (size_t)NNODES * DIMD);   // out@root, then m
    float* gi    = carve(d_ws, off, (size_t)NNODES * 768);
    float* gh    = carve(d_ws, off, (size_t)NNODES * 768);
    float* rdeg  = carve(d_ws, off, NNODES);
    float* gates = carve(d_ws, off, (size_t)NGRAPHS * 1024);
    float* lh    = carve(d_ws, off, (size_t)NGRAPHS * DIMD);
    float* lc    = carve(d_ws, off, (size_t)NGRAPHS * DIMD);
    float* qstar = carve(d_ws, off, (size_t)NGRAPHS * 512);
    float* escr  = carve(d_ws, off, NNODES);
    float* ascr  = carve(d_ws, off, NNODES);
    float* obuf  = carve(d_ws, off, (size_t)NGRAPHS * DIMD);
    (void)ws_size; (void)in_sizes; (void)n_in; (void)out_size;

    // fingerprint branch: fp@fc1 + b -> BN(eval) -> ELU
    wmma_gemm_bf16<0><<<dim3(4, 4), 128, 0, stream>>>(fp, fc1_w, fc1_b, fptmp,
        FPDIM, FPDIM, FPLIN, FPLIN, 0, 0);
    k_bnelu<<<64, 256, 0, stream>>>(fptmp, bn1_g, bn1_b, outfp);

    // node embed + edge MLP layer 1 + degrees
    k_lin0<<<2000, 256, 0, stream>>>(x, lin0_w, lin0_b, hbuf);
    k_edge1<<<2000, 256, 0, stream>>>(eattr, emlp_w1, emlp_b1, he);
    k_zero<<<8, 256, 0, stream>>>(rdeg, NNODES);
    k_deg<<<16, 256, 0, stream>>>(dst, rdeg);
    k_recip<<<8, 256, 0, stream>>>(rdeg);

    // 3 message-passing + GRU iterations
    for (int it = 0; it < 3; ++it) {
        k_gather<<<1000, 256, 0, stream>>>(hbuf, src, xsrc);
        // fused outer-product GEMM incl. folded b2 bias (the 67 GFLOP hot loop)
        msg_outer_gemm<<<63, 256, 0, stream>>>(xsrc, he, emlp_w2, emlp_b2, msg);
        // mean aggregate
        k_zero<<<2000, 256, 0, stream>>>(agg, NNODES * DIMD);
        k_scatter<<<4000, 256, 0, stream>>>(msg, dst, agg);
        // out @ conv_root, combine, relu
        wmma_gemm_bf16<0><<<dim3(125, 4), 128, 0, stream>>>(hbuf, conv_root, nullptr, mbuf,
            DIMD, DIMD, DIMD, DIMD, 0, 0);
        k_combine<<<2000, 256, 0, stream>>>(agg, rdeg, conv_bias, mbuf);
        // GRU gates (torch [3d, in] weights -> transB)
        wmma_gemm_bf16<1><<<dim3(125, 12), 128, 0, stream>>>(mbuf, gru_w_ih, gru_b_ih, gi,
            DIMD, DIMD, DIMD, 768, 0, 0);
        wmma_gemm_bf16<1><<<dim3(125, 12), 128, 0, stream>>>(hbuf, gru_w_hh, gru_b_hh, gh,
            DIMD, DIMD, DIMD, 768, 0, 0);
        k_gru<<<2000, 256, 0, stream>>>(gi, gh, hbuf);
    }

    // Set2Set
    k_zero<<<128, 256, 0, stream>>>(qstar, NGRAPHS * 512);
    k_zero<<<64, 256, 0, stream>>>(lh, NGRAPHS * DIMD);
    k_zero<<<64, 256, 0, stream>>>(lc, NGRAPHS * DIMD);
    for (int st = 0; st < 3; ++st) {
        wmma_gemm_bf16<1><<<dim3(4, 16), 128, 0, stream>>>(qstar, lstm_w_ih, lstm_b_ih,
            gates, 512, 512, 512, 1024, 0, 0);
        wmma_gemm_bf16<1><<<dim3(4, 16), 128, 0, stream>>>(lh, lstm_w_hh, lstm_b_hh,
            gates, DIMD, DIMD, DIMD, 1024, 1, 0);
        k_lstm<<<64, 256, 0, stream>>>(gates, lh, lc, qstar);
        k_attn_dot<<<8, 256, 0, stream>>>(hbuf, lh, batch, escr);
        k_attn_norm<<<64, 256, 0, stream>>>(escr, batch, ascr);
        k_attn_r<<<64, 256, 0, stream>>>(ascr, hbuf, batch, qstar);
    }

    // output head
    wmma_gemm_bf16<0><<<dim3(4, 4), 128, 0, stream>>>(qstar, lin1_w, lin1_b, obuf,
        512, 512, DIMD, DIMD, 0, 1);
    k_head<<<1, 128, 0, stream>>>(obuf, outfp, lin2_w, lin2_b, (float*)d_out);
}
